// GATLayer_71691594105227
// MI455X (gfx1250) — matrix operand order repaired and checked by reading, not compiled, not run on hardware
//
#include <hip/hip_runtime.h>
#include <hip/hip_bf16.h>

typedef __bf16 bf16_t;
typedef __attribute__((ext_vector_type(16))) __bf16 v16bf;
typedef __attribute__((ext_vector_type(8)))  __bf16 v8bf;
typedef __attribute__((ext_vector_type(8)))  float  v8f;

#define NN   4096   // nodes
#define INF  256    // input features
#define NH   4      // heads
#define NF   64     // features per head
#define HF   256    // NH*NF

// ---------------------------------------------------------------------------
// Phase 1: Wh = h (4096x256) @ W (256x256), ~fp32 accuracy via split-bf16
// (x = hi + lo; A*B ~= Ahi*Bhi + Alo*Bhi + Ahi*Blo), one wave = one 16x16 tile.
// Also writes transposed bf16 copy WhT[hf][n] for phase-3 B fragments.
// ---------------------------------------------------------------------------
__global__ __launch_bounds__(32)
void wh_gemm_kernel(const float* __restrict__ hmat,
                    const float* __restrict__ Wmat,
                    float* __restrict__ Wh,
                    bf16_t* __restrict__ WhT) {
  const int lane = threadIdx.x;
  const int m    = lane & 15;          // row within tile (A) / col (B,D)
  const int hi8  = (lane >> 4) << 3;   // D row offset: 0 or 8
  const int hi16 = (lane >> 4) << 4;   // B K offset: 0 or 16
  const int i0   = blockIdx.x << 4;    // node tile
  const int n0   = blockIdx.y << 4;    // output-feature tile

  v8f acc = {};
  for (int k0 = 0; k0 < INF; k0 += 32) {
    v16bf a_hi, a_lo, b_hi, b_lo;
    const float* arow = hmat + (size_t)(i0 + m) * INF + k0;
    const float* bcol = Wmat + (size_t)k0 * HF + (n0 + m);
#pragma unroll
    for (int e = 0; e < 16; ++e) {
      // A 16x32 bf16 layout: element e -> K = e + hi16
      int klA = e + hi16;
      float xa = arow[klA];
      bf16_t ah = (bf16_t)xa;
      a_hi[e] = ah;
      a_lo[e] = (bf16_t)(xa - (float)ah);
      // B 32x16 bf16 layout: element e -> K = e + hi16, N = lane&15
      int klB = e + hi16;
      float xb = bcol[(size_t)klB * HF];
      bf16_t bh = (bf16_t)xb;
      b_hi[e] = bh;
      b_lo[e] = (bf16_t)(xb - (float)bh);
    }
    acc = __builtin_amdgcn_wmma_f32_16x16x32_bf16(false, a_hi, false, b_hi,
                                                  (short)0, acc, false, false);
    acc = __builtin_amdgcn_wmma_f32_16x16x32_bf16(false, a_lo, false, b_hi,
                                                  (short)0, acc, false, false);
    acc = __builtin_amdgcn_wmma_f32_16x16x32_bf16(false, a_hi, false, b_lo,
                                                  (short)0, acc, false, false);
  }
  // D layout: element v -> row M = v + hi8, col N = lane&15
#pragma unroll
  for (int v = 0; v < 8; ++v) {
    int row = i0 + v + hi8;
    int col = n0 + m;
    float val = acc[v];
    Wh[(size_t)row * HF + col]  = val;
    WhT[(size_t)col * NN + row] = (bf16_t)val;
  }
}

// ---------------------------------------------------------------------------
// Phase 2: f1[h][n] = Wh[n,h,:].a1 ; f2[h][n] = Wh[n,h,:].a2  (head-major)
// ---------------------------------------------------------------------------
__global__ __launch_bounds__(128)
void attn_proj_kernel(const float* __restrict__ Wh,
                      const float* __restrict__ avec,
                      float* __restrict__ f1T,
                      float* __restrict__ f2T) {
  int idx = blockIdx.x * blockDim.x + threadIdx.x;
  if (idx >= NN * NH) return;
  int n  = idx >> 2;
  int hh = idx & 3;
  const float* w  = Wh + (size_t)n * HF + hh * NF;
  const float* a1 = avec;
  const float* a2 = avec + NF;
  float s1 = 0.f, s2 = 0.f;
#pragma unroll 8
  for (int f = 0; f < NF; ++f) {
    float x = w[f];
    s1 = fmaf(x, a1[f], s1);
    s2 = fmaf(x, a2[f], s2);
  }
  f1T[hh * NN + n] = s1;
  f2T[hh * NN + n] = s2;
}

// ---------------------------------------------------------------------------
// Phase 3: fused mask + leakyrelu + exp + one-pass softmax + WMMA aggregate.
// Block = 512 threads = 16 waves = 4 heads x 4 j-quarters (occupancy: 4
// waves/SIMD for latency hiding). Each wave accumulates a 16x64 partial over
// its 1024-j quarter; partials combined through LDS, waves 0..3 finalize.
//   w_ij = adj[i,j] ? exp(leakyrelu(f1[i,h] + f2[j,h])) : 0
//   out[i, h*64+f] = relu( (sum_j w_ij * Wh[j,h,f]) / (sum_j w_ij) )
// (logits are O(1), so no max-subtraction is needed for fp32 exp)
// ---------------------------------------------------------------------------
__global__ __launch_bounds__(512)
void gat_fused_kernel(const int* __restrict__ adj,
                      const bf16_t* __restrict__ WhT,   // [HF][NN]
                      const float* __restrict__ f1T,    // [NH][NN]
                      const float* __restrict__ f2T,    // [NH][NN]
                      float* __restrict__ out) {        // [NN][HF]
  __shared__ float lds_num[12 * 1024];   // partials of waves 4..15 (48 KiB)
  __shared__ float lds_den[16 * 32];

  const int i0   = blockIdx.x << 4;
  const int w    = threadIdx.x >> 5;
  const int head = w & 3;
  const int jq   = w >> 2;               // j-quarter 0..3
  const int lane = threadIdx.x & 31;
  const int m    = lane & 15;
  const int hi8  = (lane >> 4) << 3;
  const int hi16 = (lane >> 4) << 4;

  const float  f1s    = f1T[head * NN + i0 + m];
  const float* f2h    = f2T + head * NN;
  const int*   adjrow = adj + (size_t)(i0 + m) * NN;

  const bf16_t* bbase0 = WhT + (size_t)(head * NF +  0 + m) * NN;
  const bf16_t* bbase1 = WhT + (size_t)(head * NF + 16 + m) * NN;
  const bf16_t* bbase2 = WhT + (size_t)(head * NF + 32 + m) * NN;
  const bf16_t* bbase3 = WhT + (size_t)(head * NF + 48 + m) * NN;

  v8f c0 = {}, c1 = {}, c2 = {}, c3 = {};
  float den = 0.f;

  const int jbeg = jq << 10;
  const int jend = jbeg + 1024;
  for (int j0 = jbeg; j0 < jend; j0 += 32) {
    // Pull the adj stream (HBM-resident, read exactly once) ahead of use.
    if (j0 + 288 <= NN)                          // uniform scalar branch
      __builtin_prefetch(adjrow + j0 + 256 + hi16, 0, 1);  // global_prefetch_b8

    // ---- A fragment: attention weights, 16x32 bf16 (contiguous K layout)
    v16bf afrag;
#pragma unroll
    for (int r = 0; r < 2; ++r) {
      int kb = j0 + r * 8 + hi16;                // 8 consecutive j's
      int4   av0 = *(const int4*)  (adjrow + kb);
      int4   av1 = *(const int4*)  (adjrow + kb + 4);
      float4 fv0 = *(const float4*)(f2h    + kb);
      float4 fv1 = *(const float4*)(f2h    + kb + 4);
      int   adjv[8] = {av0.x, av0.y, av0.z, av0.w, av1.x, av1.y, av1.z, av1.w};
      float f2v[8]  = {fv0.x, fv0.y, fv0.z, fv0.w, fv1.x, fv1.y, fv1.z, fv1.w};
#pragma unroll
      for (int e = 0; e < 8; ++e) {
        float t = f1s + f2v[e];
        t = (t > 0.f) ? t : 0.2f * t;            // leakyrelu(0.2)
        float wgt = adjv[e] ? __expf(t) : 0.f;   // mask -> exp(-1e9) == 0
        den += wgt;
        afrag[r * 8 + e] = (bf16_t)wgt;
      }
    }
    // ---- B fragments: Wh columns (contiguous in transposed bf16 copy)
#pragma unroll
    for (int t = 0; t < 4; ++t) {
      const bf16_t* bp = (t == 0) ? bbase0 : (t == 1) ? bbase1
                       : (t == 2) ? bbase2 : bbase3;
      v8bf q0 = *(const v8bf*)(bp + j0 + hi16);
      v8bf q1 = *(const v8bf*)(bp + j0 + hi16 + 8);
      v16bf bfrag;
#pragma unroll
      for (int e = 0; e < 8; ++e) { bfrag[e] = q0[e]; bfrag[8 + e] = q1[e]; }
      if      (t == 0) c0 = __builtin_amdgcn_wmma_f32_16x16x32_bf16(false, afrag, false, bfrag, (short)0, c0, false, false);
      else if (t == 1) c1 = __builtin_amdgcn_wmma_f32_16x16x32_bf16(false, afrag, false, bfrag, (short)0, c1, false, false);
      else if (t == 2) c2 = __builtin_amdgcn_wmma_f32_16x16x32_bf16(false, afrag, false, bfrag, (short)0, c2, false, false);
      else             c3 = __builtin_amdgcn_wmma_f32_16x16x32_bf16(false, afrag, false, bfrag, (short)0, c3, false, false);
    }
  }

  // lanes L and L^16 hold complementary K-halves of row m -> quarter row sum:
  den += __shfl_xor(den, 16, 32);                // every lane: den_q[lane&15]

  float creg[32];
#pragma unroll
  for (int v = 0; v < 8; ++v) {
    creg[v]      = c0[v];
    creg[8 + v]  = c1[v];
    creg[16 + v] = c2[v];
    creg[24 + v] = c3[v];
  }

  // ---- combine the 4 j-quarter partials per head through LDS
  if (w >= 4) {
    float* dst = lds_num + (w - 4) * 1024;
#pragma unroll
    for (int idx = 0; idx < 32; ++idx)
      dst[idx * 32 + lane] = creg[idx];          // conflict-free
  }
  lds_den[w * 32 + lane] = den;
  __syncthreads();

  if (w < 4) {
    den += lds_den[(w + 4) * 32 + lane]
         + lds_den[(w + 8) * 32 + lane]
         + lds_den[(w + 12) * 32 + lane];
    const float* p0 = lds_num + (w + 0) * 1024;  // wave w+4
    const float* p1 = lds_num + (w + 4) * 1024;  // wave w+8
    const float* p2 = lds_num + (w + 8) * 1024;  // wave w+12
#pragma unroll
    for (int idx = 0; idx < 32; ++idx)
      creg[idx] += p0[idx * 32 + lane] + p1[idx * 32 + lane] + p2[idx * 32 + lane];

#pragma unroll
    for (int v = 0; v < 8; ++v) {
      int M = v + hi8;                           // output row within tile
      float dv  = __shfl(den, M, 32);            // den[M] lives in lane M
      float inv = 1.0f / dv;
      float* orow = out + (size_t)(i0 + M) * HF + head * NF + m;
#pragma unroll
      for (int t = 0; t < 4; ++t) {
        float o = creg[t * 8 + v] * inv;
        orow[t * 16] = (o > 0.f) ? o : 0.f;      // final relu
      }
    }
  }
}

// ---------------------------------------------------------------------------
extern "C" void kernel_launch(void* const* d_in, const int* in_sizes, int n_in,
                              void* d_out, int out_size, void* d_ws, size_t ws_size,
                              hipStream_t stream) {
  (void)in_sizes; (void)n_in; (void)out_size; (void)ws_size;
  const float* hmat = (const float*)d_in[0];   // (4096, 256) fp32
  const int*   adj  = (const int*)  d_in[1];   // (4096, 4096) int32
  const float* Wmat = (const float*)d_in[2];   // (256, 256) fp32
  const float* avec = (const float*)d_in[3];   // (128, 1) fp32
  float* out = (float*)d_out;                  // (4096, 256) fp32

  char* ws = (char*)d_ws;
  float*  Wh  = (float*) (ws + 0);                      // 4096*256*4  = 4 MiB
  bf16_t* WhT = (bf16_t*)(ws + (size_t)4 * 1024 * 1024);// 256*4096*2  = 2 MiB
  float*  f1T = (float*) (ws + (size_t)6 * 1024 * 1024);// 4*4096*4    = 64 KiB
  float*  f2T = (float*) (ws + (size_t)6 * 1024 * 1024 + 64 * 1024);

  wh_gemm_kernel <<<dim3(NN / 16, HF / 16), 32, 0, stream>>>(hmat, Wmat, Wh, WhT);
  attn_proj_kernel<<<(NN * NH + 127) / 128, 128, 0, stream>>>(Wh, avec, f1T, f2T);
  gat_fused_kernel<<<NN / 16, 512, 0, stream>>>(adj, WhT, f1T, f2T, out);
}
